// CAM_Module_5609227289232
// MI455X (gfx1250) — matrix operand order repaired and checked by reading, not compiled, not run on hardware
//
#include <hip/hip_runtime.h>
#include <hip/hip_bf16.h>

// ---- problem constants (from reference) ----
#define Bsz  32
#define Cch  2048
#define Nsp  196          // H*W
#define NP   224          // spatial K padded to multiple of 32
#define KK   7            // NP / 32
#define NT   13           // ceil(196/16) output n-tiles
#define NPAD (NT * 16)    // 208
#define OC   512

// LDS row pitches chosen for conflict-free 16-lane row-strided reads
#define QDP  232          // q rows (stride 464B = 116 dw; 116%64=52, gcd(52,64)=4 -> ok)
#define QTP  40           // qT d-window rows (stride 80B = 20 dw; distinct banks)
#define PAP  40           // P staging rows
#define CVP  2056         // conv B panel rows (stride 4112B = 1028 dw; 1028%64=4 -> ok)

typedef __attribute__((ext_vector_type(16))) __bf16 v16bf;
typedef __attribute__((ext_vector_type(8)))  float  v8f;
typedef __attribute__((ext_vector_type(4)))  int    vi4;

union Frag { int4 i[2]; v16bf v; };   // 32 bytes: one bf16 WMMA A/B fragment per lane

__device__ __forceinline__ v8f wmma_bf16(v16bf a, v16bf b, v8f c) {
  // D = A(16x32 bf16) * B(32x16 bf16) + C(16x16 f32)
  return __builtin_amdgcn_wmma_f32_16x16x32_bf16(false, a, false, b, (short)0, c,
                                                 false, false);
}

__device__ __forceinline__ unsigned short f2bf(float f) {   // RNE f32 -> bf16
  union { float f; unsigned u; } v; v.f = f;
  unsigned r = v.u + 0x7FFFu + ((v.u >> 16) & 1u);
  return (unsigned short)(r >> 16);
}

#define V8F_ZERO {0.f,0.f,0.f,0.f,0.f,0.f,0.f,0.f}

// CDNA5 async global->LDS staging, if the toolchain exposes the builtins.
#if defined(__has_builtin)
#  if __has_builtin(__builtin_amdgcn_global_load_async_to_lds_b128) && \
      __has_builtin(__builtin_amdgcn_s_wait_asynccnt)
#    define USE_ASYNC_LDS 1
#  endif
#endif
#ifndef USE_ASYNC_LDS
#  define USE_ASYNC_LDS 0
#endif

#define AS1_ __attribute__((address_space(1)))
#define AS3_ __attribute__((address_space(3)))

__device__ __forceinline__ void async_cp16(const unsigned short* g, unsigned short* l) {
#if USE_ASYNC_LDS
  __builtin_amdgcn_global_load_async_to_lds_b128(
      (AS1_ vi4*)(void*)g, (AS3_ vi4*)(void*)l, 0, 0);
#else
  *(int4*)l = *(const int4*)g;
#endif
}
__device__ __forceinline__ void async_wait_all() {
#if USE_ASYNC_LDS
  __builtin_amdgcn_s_wait_asynccnt(0);
#endif
}

// ---------------------------------------------------------------------------
// Pack kernels: fp32 -> bf16 workspace operands
// ---------------------------------------------------------------------------
__global__ void pack_q_kernel(const float* __restrict__ x, unsigned short* __restrict__ q) {
  long idx = (long)blockIdx.x * blockDim.x + threadIdx.x;
  long total = (long)Bsz * Cch * NP;
  if (idx >= total) return;
  int k = (int)(idx % NP);
  long row = idx / NP;
  float v = (k < Nsp) ? x[row * Nsp + k] : 0.0f;
  q[idx] = f2bf(v);
}

__global__ void pack_qT_kernel(const float* __restrict__ x, unsigned short* __restrict__ qT) {
  // qT[b][n][c] = x[b][c][n], n padded to NPAD with zeros; LDS tile transpose
  __shared__ float tile[16][17];
  int b  = blockIdx.z;
  int c0 = blockIdx.y * 16;
  int n0 = blockIdx.x * 16;
  int tx = threadIdx.x, ty = threadIdx.y;
  int n = n0 + tx;
  float v = (n < Nsp) ? x[((long)b * Cch + (c0 + ty)) * Nsp + n] : 0.0f;
  tile[ty][tx] = v;
  __syncthreads();
  qT[((long)b * NPAD + (n0 + ty)) * Cch + (c0 + tx)] = f2bf(tile[tx][ty]);
}

__global__ void pack_w_kernel(const float* __restrict__ w, unsigned short* __restrict__ wbf) {
  int idx = blockIdx.x * blockDim.x + threadIdx.x;
  if (idx < OC * Cch) wbf[idx] = f2bf(w[idx]);
}

// ---------------------------------------------------------------------------
// Cooperative double-buffered staging of one 32-wide d-window:
//   qd : q rows d0..d0+31 (all padded K)      -> energy B operand
//   qtd: qT rows 0..207, cols d0..d0+31      -> AV B operand
// Shared by all 4 waves of the block (same batch, same d schedule).
// ---------------------------------------------------------------------------
__device__ __forceinline__ void stage_dblock(int tid,
    const unsigned short* qb, const unsigned short* qTb, int d0,
    unsigned short (*qdb)[QDP], unsigned short (*qtb)[QTP]) {
  for (int i = tid; i < 32 * (NP / 8); i += 128) {        // 896 x 16B
    int j = i / (NP / 8), col = (i % (NP / 8)) * 8;
    async_cp16(qb + (long)(d0 + j) * NP + col, &qdb[j][col]);
  }
  for (int i = tid; i < NPAD * 4; i += 128) {             // 832 x 16B
    int n = i / 4, col = (i % 4) * 8;
    async_cp16(qTb + (long)n * Cch + d0 + col, &qtb[n][col]);
  }
}

// ---------------------------------------------------------------------------
// Fused flash-style CAM: single pass over d with online (min, sum) softmax.
// One wave owns a 16-row channel tile; the block shares staged B operands.
// ---------------------------------------------------------------------------
__global__ void __launch_bounds__(128)
cam_fused_kernel(const unsigned short* __restrict__ q,
                 const unsigned short* __restrict__ qT,
                 const float* __restrict__ x,
                 const float* __restrict__ gamma,
                 unsigned short* __restrict__ outT) {
  __shared__ unsigned short ldsA[4][16][QDP];    // per-wave q c-rows (A operand)
  __shared__ unsigned short pa[4][16][PAP];      // per-wave P staging
  __shared__ unsigned short qd[2][32][QDP];      // shared energy-B, double buffered
  __shared__ unsigned short qtd[2][NPAD][QTP];   // shared AV-B, double buffered

  int tid = threadIdx.x;
  int wave = tid >> 5, lane = tid & 31;
  int half = lane >> 4, lq = lane & 15;
  int gw = blockIdx.x * 4 + wave;
  int b  = gw >> 7;                 // C/16 = 128 c-tiles per batch
  int c0 = (gw & 127) << 4;
  const unsigned short* qb  = q  + (long)b * Cch * NP;
  const unsigned short* qTb = qT + (long)b * NPAD * Cch;

  // per-wave A rows (c0..c0+15, full padded K)
  for (int i = lane; i < 16 * (NP / 8); i += 32) {
    int row = i / (NP / 8), col = (i % (NP / 8)) * 8;
    async_cp16(qb + (long)(c0 + row) * NP + col, &ldsA[wave][row][col]);
  }
  // first d-window
  stage_dblock(tid, qb, qTb, 0, qd[0], qtd[0]);
  async_wait_all();
  __syncthreads();

  float m[8], s[8];
#pragma unroll
  for (int r = 0; r < 8; ++r) { m[r] = 3.0e38f; s[r] = 0.0f; }

  v8f acc[NT];
#pragma unroll
  for (int nt = 0; nt < NT; ++nt) acc[nt] = (v8f)V8F_ZERO;

  for (int d0 = 0; d0 < Cch; d0 += 32) {
    int buf = (d0 >> 5) & 1;
    if (d0 + 32 < Cch)                       // overlap next window's fill
      stage_dblock(tid, qb, qTb, d0 + 32, qd[buf ^ 1], qtd[buf ^ 1]);

    // ---- two 16-wide energy tiles (E = q c-rows x q d-rows^T), all from LDS ----
    v8f e0 = V8F_ZERO, e1 = V8F_ZERO;
#pragma unroll
    for (int kk = 0; kk < KK; ++kk) {
      Frag Af, B0, B1;
      Af.i[0] = *(const int4*)&ldsA[wave][lq][kk * 32 + half * 8];
      Af.i[1] = *(const int4*)&ldsA[wave][lq][kk * 32 + 16 + half * 8];
      B0.i[0] = *(const int4*)&qd[buf][lq][kk * 32 + half * 8];
      B0.i[1] = *(const int4*)&qd[buf][lq][kk * 32 + 16 + half * 8];
      B1.i[0] = *(const int4*)&qd[buf][16 + lq][kk * 32 + half * 8];
      B1.i[1] = *(const int4*)&qd[buf][16 + lq][kk * 32 + 16 + half * 8];
      e0 = wmma_bf16(Af.v, B0.v, e0);
      e1 = wmma_bf16(Af.v, B1.v, e1);
    }
    // ---- online softmax update (row r lives in the same 16-lane group) ----
#pragma unroll
    for (int r = 0; r < 8; ++r) {
      float tm = fminf(e0[r], e1[r]);
      tm = fminf(tm, __shfl_xor(tm, 1, 32));
      tm = fminf(tm, __shfl_xor(tm, 2, 32));
      tm = fminf(tm, __shfl_xor(tm, 4, 32));
      tm = fminf(tm, __shfl_xor(tm, 8, 32));
      float newm  = fminf(m[r], tm);
      float scale = __expf(newm - m[r]);
      float p0 = __expf(newm - e0[r]);
      float p1 = __expf(newm - e1[r]);
      pa[wave][half * 8 + r][lq]      = f2bf(p0);
      pa[wave][half * 8 + r][16 + lq] = f2bf(p1);
      float ps = p0 + p1;
      ps += __shfl_xor(ps, 1, 32);
      ps += __shfl_xor(ps, 2, 32);
      ps += __shfl_xor(ps, 4, 32);
      ps += __shfl_xor(ps, 8, 32);
      s[r] = s[r] * scale + ps;
      m[r] = newm;
#pragma unroll
      for (int nt = 0; nt < NT; ++nt) acc[nt][r] *= scale;   // flash rescale
    }
    // ---- reload P as an A-fragment (same-wave LDS ops are in-order) ----
    Frag P;
    P.i[0] = *(const int4*)&pa[wave][lq][half * 8];
    P.i[1] = *(const int4*)&pa[wave][lq][16 + half * 8];

#pragma unroll
    for (int nt = 0; nt < NT; ++nt) {
      Frag Bf;
      Bf.i[0] = *(const int4*)&qtd[buf][nt * 16 + lq][half * 8];
      Bf.i[1] = *(const int4*)&qtd[buf][nt * 16 + lq][16 + half * 8];
      acc[nt] = wmma_bf16(P.v, Bf.v, acc[nt]);
    }
    async_wait_all();    // our share of next window landed
    __syncthreads();     // whole block done reading buf / staging buf^1
  }

  // ---- epilogue: out = gamma * (acc/s) + x, store bf16 outT[b][n][c] ----
  float inv[8];
#pragma unroll
  for (int r = 0; r < 8; ++r) inv[r] = 1.0f / s[r];
  float g = gamma[0];
  const float* xb = x + ((long)b * Cch + c0 + half * 8) * Nsp;
  unsigned short* ob = outT + (long)b * NPAD * Cch + c0 + half * 8;

#pragma unroll
  for (int nt = 0; nt < NT; ++nt) {
    int n = nt * 16 + lq;
    if (n < Nsp) {
      union { unsigned short u[8]; int4 v; } pk;
#pragma unroll
      for (int r = 0; r < 8; ++r) {
        float att = acc[nt][r] * inv[r];
        float o = g * att + xb[(long)r * Nsp + n];
        pk.u[r] = f2bf(o);
      }
      *(int4*)(ob + (long)n * Cch) = pk.v;   // outT[b][n][c0+half*8 .. +7]
    }
  }
}

// ---------------------------------------------------------------------------
// 1x1 conv as GEMM  D[o,n] = sum_c W[o,c] * out[c,n]  (+ bias)
// Block owns (b, n-tile); 4 waves cover 4 o-tiles and share the outT panel,
// async-staged into LDS once (16 rows x 2048 c = 64KB).
// ---------------------------------------------------------------------------
__global__ void __launch_bounds__(128)
cam_conv_kernel(const unsigned short* __restrict__ wbf,
                const unsigned short* __restrict__ outT,
                const float* __restrict__ bias,
                float* __restrict__ out) {
  __shared__ unsigned short bs[16][CVP];

  int tid = threadIdx.x;
  int wave = tid >> 5, lane = tid & 31;
  int half = lane >> 4, lq = lane & 15;
  int blk = blockIdx.x;                 // B * NT * 8 blocks
  int b   = blk / (NT * 8);
  int rem = blk % (NT * 8);
  int nt  = rem / 8, og = rem % 8;
  int ot  = og * 4 + wave;

  // stage the outT n-panel (shared B operand) into LDS
  const unsigned short* src = outT + ((long)b * NPAD + nt * 16) * Cch;
  for (int i = tid; i < 16 * (Cch / 8); i += 128) {       // 4096 x 16B
    int rrow = i / (Cch / 8), col = (i % (Cch / 8)) * 8;
    async_cp16(src + (long)rrow * Cch + col, &bs[rrow][col]);
  }
  async_wait_all();
  __syncthreads();

  const unsigned short* wrow = wbf + (long)(ot * 16 + lq) * Cch;
  v8f acc = V8F_ZERO;
  for (int c0 = 0; c0 < Cch; c0 += 32) {
    Frag Af, Bf;
    Af.i[0] = *(const int4*)(wrow + c0 + half * 8);
    Af.i[1] = *(const int4*)(wrow + c0 + 16 + half * 8);
    Bf.i[0] = *(const int4*)&bs[lq][c0 + half * 8];
    Bf.i[1] = *(const int4*)&bs[lq][c0 + 16 + half * 8];
    acc = wmma_bf16(Af.v, Bf.v, acc);
    if (c0 + 32 < Cch) __builtin_prefetch(wrow + c0 + 64, 0, 3);
  }

  int n = nt * 16 + lq;
  if (n < Nsp) {
    const float* bp = bias + ot * 16 + half * 8;
    float* op = out + ((long)b * OC + ot * 16 + half * 8) * Nsp + n;
#pragma unroll
    for (int r = 0; r < 8; ++r) op[(long)r * Nsp] = acc[r] + bp[r];
  }
}

// ---------------------------------------------------------------------------
extern "C" void kernel_launch(void* const* d_in, const int* in_sizes, int n_in,
                              void* d_out, int out_size, void* d_ws, size_t ws_size,
                              hipStream_t stream) {
  (void)in_sizes; (void)n_in; (void)out_size;
  const float* x      = (const float*)d_in[0];
  const float* gamma  = (const float*)d_in[1];
  const float* conv_w = (const float*)d_in[2];
  const float* conv_b = (const float*)d_in[3];
  float* out = (float*)d_out;
  char* ws = (char*)d_ws;

  size_t off_q  = 0;
  size_t off_qT = off_q  + (size_t)Bsz * Cch * NP * 2;      // q  bf16 [B][C][NP]
  size_t off_w  = off_qT + (size_t)Bsz * NPAD * Cch * 2;    // qT bf16 [B][NPAD][C]
  size_t off_oT = off_w  + (size_t)OC * Cch * 2;            // W  bf16 [OC][C]
  size_t need   = off_oT + (size_t)Bsz * NPAD * Cch * 2;    // outT bf16 [B][NPAD][C]
  if (ws_size < need) return;

  unsigned short* q   = (unsigned short*)(ws + off_q);
  unsigned short* qT  = (unsigned short*)(ws + off_qT);
  unsigned short* wb  = (unsigned short*)(ws + off_w);
  unsigned short* oT  = (unsigned short*)(ws + off_oT);

  { long total = (long)Bsz * Cch * NP;
    pack_q_kernel<<<(int)((total + 255) / 256), 256, 0, stream>>>(x, q); }
  { dim3 g(NT, Cch / 16, Bsz), bl(16, 16);
    pack_qT_kernel<<<g, bl, 0, stream>>>(x, qT); }
  { int total = OC * Cch;
    pack_w_kernel<<<(total + 255) / 256, 256, 0, stream>>>(conv_w, wb); }

  cam_fused_kernel<<<(Bsz * Cch / 16) / 4, 128, 0, stream>>>(q, qT, x, gamma, oT);
  cam_conv_kernel<<<Bsz * NT * 8, 128, 0, stream>>>(wb, oT, conv_b, out);
}
